// OuterProductMean_33268816675078
// MI455X (gfx1250) — matrix operand order repaired and checked by reading, not compiled, not run on hardware
//
#include <hip/hip_runtime.h>
#include <hip/hip_bf16.h>

// ---------------------------------------------------------------------------
// OuterProductMean (AlphaFold) for MI455X / gfx1250, bf16 WMMA with f32 acc.
// S=512, R=384, C_MSA=64, C_OUTER=32, C_OUT=128.
// Round 2: async-to-LDS staging of the shared A panel (ASYNCcnt path) +
// register software-pipelining of B fragments so WMMA issue is not gated by
// s_wait_loadcnt 0 on every k-step.
// ---------------------------------------------------------------------------

#define S_DIM   512
#define R_DIM   384
#define CMSA    64
#define COUTER  32
#define COUT    128

typedef __bf16 bf16;
typedef __attribute__((ext_vector_type(16))) __bf16        v16bf;
typedef __attribute__((ext_vector_type(8)))  float         v8f;
typedef __attribute__((ext_vector_type(4)))  unsigned int  u32x4;

union Frag16 { v16bf v; u32x4 q[2]; };

// ---- CDNA5 async global->LDS copy (ASYNCcnt), inline asm -------------------
__device__ __forceinline__ void async_g2l_b128(unsigned lds_off, const void* gaddr) {
    // GV mode: LDS byte address in a VGPR (dest), 64-bit global address VGPR pair.
    asm volatile("global_load_async_to_lds_b128 %0, %1, off"
                 :: "v"(lds_off), "v"(gaddr) : "memory");
}
__device__ __forceinline__ void wait_asynccnt0() {
    asm volatile("s_wait_asynccnt 0x0" ::: "memory");
}

// ---------------------------------------------------------------------------
// Kernel 1: LayerNorm + masked projections, store transposed bf16:
//   leftT[(r*32 + o)*512 + s], rightT likewise  (K = s contiguous)
// One wave (32 lanes) per (s,r) row; 8 rows per 256-thread block.
// ---------------------------------------------------------------------------
__global__ __launch_bounds__(256)
void opm_ln_proj_kernel(const float* __restrict__ msa,
                        const float* __restrict__ mask,
                        const float* __restrict__ gamma,
                        const float* __restrict__ beta,
                        const float* __restrict__ w_left,
                        const float* __restrict__ w_right,
                        bf16* __restrict__ leftT,
                        bf16* __restrict__ rightT) {
    __shared__ float sWL[COUTER * CMSA];
    __shared__ float sWR[COUTER * CMSA];
    for (int i = threadIdx.x; i < COUTER * CMSA; i += blockDim.x) {
        sWL[i] = w_left[i];
        sWR[i] = w_right[i];
    }
    __syncthreads();

    const int lane = threadIdx.x & 31;
    const int wave = threadIdx.x >> 5;
    const int row  = blockIdx.x * 8 + wave;      // row = s*R + r
    const int s    = row / R_DIM;
    const int r    = row % R_DIM;

    const float* mrow = msa + (size_t)row * CMSA;
    float v0 = mrow[lane];
    float v1 = mrow[lane + 32];

    // wave32 reductions for mean / variance
    float sum = v0 + v1;
    for (int off = 16; off > 0; off >>= 1) sum += __shfl_xor(sum, off, 32);
    float mu = sum * (1.0f / 64.0f);
    float d0 = v0 - mu, d1 = v1 - mu;
    float sq = d0 * d0 + d1 * d1;
    for (int off = 16; off > 0; off >>= 1) sq += __shfl_xor(sq, off, 32);
    float rstd = rsqrtf(sq * (1.0f / 64.0f) + 1e-5f);

    float x0 = d0 * rstd * gamma[lane]      + beta[lane];
    float x1 = d1 * rstd * gamma[lane + 32] + beta[lane + 32];

    // lane o computes output channel o for both projections; x broadcast via shfl
    const int o = lane;
    float accL = 0.f, accR = 0.f;
    for (int k = 0; k < 32; ++k) {
        float xa = __shfl(x0, k, 32);
        float xb = __shfl(x1, k, 32);
        accL += xa * sWL[o * CMSA + k] + xb * sWL[o * CMSA + k + 32];
        accR += xa * sWR[o * CMSA + k] + xb * sWR[o * CMSA + k + 32];
    }
    float m = mask[(size_t)s * R_DIM + r];
    accL *= m;
    accR *= m;

    size_t idx = ((size_t)(r * COUTER + o)) * S_DIM + s;
    leftT[idx]  = (bf16)accL;
    rightT[idx] = (bf16)accR;
}

// ---------------------------------------------------------------------------
// Kernel 2: transpose output_w [c][e][f] f32 -> WT [f][c*32+e] bf16
// ---------------------------------------------------------------------------
__global__ void opm_pack_w_kernel(const float* __restrict__ output_w,
                                  bf16* __restrict__ WT) {
    int t = blockIdx.x * blockDim.x + threadIdx.x;   // over 32*32*128
    if (t >= COUTER * COUTER * COUT) return;
    int f  = t % COUT;
    int ce = t / COUT;
    WT[(size_t)f * (COUTER * COUTER) + ce] = (bf16)output_w[t];
}

// ---------------------------------------------------------------------------
// Kernel 3: norm[b,d] = sum_s mask[s,b]*mask[s,d]
// ---------------------------------------------------------------------------
__global__ void opm_norm_kernel(const float* __restrict__ mask,
                                float* __restrict__ norm) {
    int t = blockIdx.x * blockDim.x + threadIdx.x;
    if (t >= R_DIM * R_DIM) return;
    int b = t / R_DIM, d = t % R_DIM;
    float acc = 0.f;
    for (int s = 0; s < S_DIM; ++s)
        acc += mask[s * R_DIM + b] * mask[s * R_DIM + d];
    norm[t] = acc;
}

// ---------------------------------------------------------------------------
// Kernel 4 (fused, WMMA): per workgroup: one b, 16 d's.
//   Stage leftT b-panel (32x512 bf16, 32 KB) into LDS via async copies.
//   GEMM1: outer[c(32), d_loc*32+e(512)] = A(LDS) x rightT  (K=512), B frags
//          double-buffered in registers.
//   -> bf16 in LDS [16 d][1024 ce] -> GEMM2 with WT (K=1024, N=128 f)
//   -> + bias, / (eps + norm), store f32 out[b][d][f].
// ---------------------------------------------------------------------------
__global__ __launch_bounds__(256)
void opm_fused_kernel(const bf16* __restrict__ leftT,
                      const bf16* __restrict__ rightT,
                      const bf16* __restrict__ WT,
                      const float* __restrict__ norm,
                      const float* __restrict__ bias,
                      float* __restrict__ out) {
    __shared__ __align__(16) bf16 sA[COUTER * S_DIM];   // 32 KB: A panel [c][s]
    __shared__ __align__(16) bf16 sOuter[16 * 1024];    // 32 KB: [d_local][c*32+e]

    const int tid   = threadIdx.x;
    const int lane  = tid & 31;
    const int wave  = tid >> 5;
    const int b     = blockIdx.y;
    const int d0    = blockIdx.x * 16;

    const int halfk = lane >> 4;    // which K half this lane holds
    const int mN    = lane & 15;    // A row / B column within a tile

    const v8f vzero = {0.f, 0.f, 0.f, 0.f, 0.f, 0.f, 0.f, 0.f};

    // ---- Stage A panel into LDS with async copies (ASYNCcnt path) ---------
    {
        const bf16* Abase = leftT + (size_t)(b * COUTER) * S_DIM;
        const unsigned lds_base = (unsigned)(size_t)(const void*)sA;
        // 32*512 bf16 = 2048 x 16B chunks; 8 chunks per thread
        for (int c = tid; c < (COUTER * S_DIM) / 8; c += 256)
            async_g2l_b128(lds_base + (unsigned)c * 16u, Abase + c * 8);
        wait_asynccnt0();
        __syncthreads();
    }

    // ---------------- GEMM1 ------------------------------------------------
    // Wave w owns n-tiles j = 4w..4w+3 (n = d_local*32 + e), both c-tiles.
    v8f acc[2][4];
    for (int mt = 0; mt < 2; ++mt)
        for (int jj = 0; jj < 4; ++jj) acc[mt][jj] = vzero;

    auto loadAf = [&](int kt, Frag16 af[2]) {
        // A frag (16x32 bf16): lane half -> 16 contiguous K
        const int k0a = kt * 32 + halfk * 16;
        for (int mt = 0; mt < 2; ++mt) {
            const bf16* ap = &sA[(mt * 16 + mN) * S_DIM + k0a];
            af[mt].q[0] = *reinterpret_cast<const u32x4*>(ap);
            af[mt].q[1] = *reinterpret_cast<const u32x4*>(ap + 8);
        }
    };
    auto loadBf = [&](int kt, Frag16 bfr[4]) {
        // B frag (32x16 bf16): lane half -> 16 contiguous K
        const int k0b = kt * 32 + halfk * 16;
        for (int jj = 0; jj < 4; ++jj) {
            const int j       = wave * 4 + jj;
            const int d_local = j >> 1;
            const int e0      = (j & 1) * 16;
            const bf16* bp = rightT
                + (size_t)((d0 + d_local) * COUTER + e0 + mN) * S_DIM + k0b;
            bfr[jj].q[0] = *reinterpret_cast<const u32x4*>(bp);
            bfr[jj].q[1] = *reinterpret_cast<const u32x4*>(bp + 8);
        }
    };
    auto mma8 = [&](Frag16 af[2], Frag16 bfr[4]) {
        for (int jj = 0; jj < 4; ++jj)
            for (int mt = 0; mt < 2; ++mt)
                acc[mt][jj] = __builtin_amdgcn_wmma_f32_16x16x32_bf16(
                    false, af[mt].v, false, bfr[jj].v, (short)0, acc[mt][jj],
                    false, false);
    };

    {
        Frag16 aF[2], bPing[4], bPong[4];
        loadBf(0, bPing);                      // prologue
        for (int kt = 0; kt < S_DIM / 32; kt += 2) {
            loadBf(kt + 1, bPong);             // in flight during mma on bPing
            loadAf(kt, aF);
            mma8(aF, bPing);
            if (kt + 2 < S_DIM / 32) loadBf(kt + 2, bPing);
            loadAf(kt + 1, aF);
            mma8(aF, bPong);
        }
    }

    // Spill outer tile (f32 C layout -> bf16 LDS [d_local][c*32+e]).
    // C layout: VGPR rr, lanes 0-15: M=rr, N=lane; lanes 16-31: M=rr+8.
    for (int mt = 0; mt < 2; ++mt)
        for (int jj = 0; jj < 4; ++jj) {
            const int j       = wave * 4 + jj;
            const int d_local = j >> 1;
            const int e       = (j & 1) * 16 + mN;
            for (int rr = 0; rr < 8; ++rr) {
                const int c = mt * 16 + rr + halfk * 8;
                sOuter[d_local * 1024 + c * 32 + e] = (bf16)acc[mt][jj][rr];
            }
        }
    __syncthreads();

    // ---------------- GEMM2 ------------------------------------------------
    // M = 16 d's, K = 1024 (c,e), N = 128 f; wave w owns f-tile [16w, 16w+16).
    v8f acc2 = vzero;
    const int f0 = wave * 16;

    auto loadA2 = [&](int kt, Frag16& a2) {
        const int k0a = kt * 32 + halfk * 16;
        const bf16* ap = &sOuter[mN * 1024 + k0a];
        a2.q[0] = *reinterpret_cast<const u32x4*>(ap);
        a2.q[1] = *reinterpret_cast<const u32x4*>(ap + 8);
    };
    auto loadW = [&](int kt, Frag16& w) {
        const int k0b = kt * 32 + halfk * 16;
        const bf16* bp = WT + (size_t)(f0 + mN) * 1024 + k0b;
        w.q[0] = *reinterpret_cast<const u32x4*>(bp);
        w.q[1] = *reinterpret_cast<const u32x4*>(bp + 8);
    };

    {
        Frag16 a2, wPing, wPong;
        loadW(0, wPing);
        for (int kt = 0; kt < 1024 / 32; kt += 2) {
            loadW(kt + 1, wPong);
            loadA2(kt, a2);
            acc2 = __builtin_amdgcn_wmma_f32_16x16x32_bf16(
                false, a2.v, false, wPing.v, (short)0, acc2, false, false);
            if (kt + 2 < 1024 / 32) loadW(kt + 2, wPing);
            loadA2(kt + 1, a2);
            acc2 = __builtin_amdgcn_wmma_f32_16x16x32_bf16(
                false, a2.v, false, wPong.v, (short)0, acc2, false, false);
        }
    }

    // Epilogue: + bias[f], / (1e-3 + norm[b,d]); store f32.
    const int f  = f0 + mN;
    const float bf = bias[f];
    for (int rr = 0; rr < 8; ++rr) {
        const int d_local = rr + halfk * 8;
        const int d = d0 + d_local;
        const float nv = norm[b * R_DIM + d];
        out[((size_t)b * R_DIM + d) * COUT + f] = (acc2[rr] + bf) / (1e-3f + nv);
    }
}

// ---------------------------------------------------------------------------
// Host launcher
// ---------------------------------------------------------------------------
extern "C" void kernel_launch(void* const* d_in, const int* in_sizes, int n_in,
                              void* d_out, int out_size, void* d_ws, size_t ws_size,
                              hipStream_t stream) {
    (void)in_sizes; (void)n_in; (void)out_size; (void)ws_size;

    const float* msa      = (const float*)d_in[0];
    const float* mask     = (const float*)d_in[1];
    const float* ln_gamma = (const float*)d_in[2];
    const float* ln_beta  = (const float*)d_in[3];
    const float* w_left   = (const float*)d_in[4];
    const float* w_right  = (const float*)d_in[5];
    const float* output_w = (const float*)d_in[6];
    const float* output_b = (const float*)d_in[7];
    float* out = (float*)d_out;

    // Workspace layout (bytes):
    //   leftT  bf16 [R*32][S]  : 12,582,912
    //   rightT bf16 [R*32][S]  : 12,582,912
    //   WT     bf16 [128][1024]:    262,144
    //   norm   f32  [R][R]     :    589,824
    char* ws = (char*)d_ws;
    bf16*  leftT  = (bf16*)(ws);
    bf16*  rightT = (bf16*)(ws + 12582912);
    bf16*  WT     = (bf16*)(ws + 25165824);
    float* nrm    = (float*)(ws + 25427968);

    opm_ln_proj_kernel<<<dim3((S_DIM * R_DIM) / 8), 256, 0, stream>>>(
        msa, mask, ln_gamma, ln_beta, w_left, w_right, leftT, rightT);

    opm_pack_w_kernel<<<dim3((COUTER * COUTER * COUT + 255) / 256), 256, 0, stream>>>(
        output_w, WT);

    opm_norm_kernel<<<dim3((R_DIM * R_DIM + 255) / 256), 256, 0, stream>>>(
        mask, nrm);

    opm_fused_kernel<<<dim3(R_DIM / 16, R_DIM), 256, 0, stream>>>(
        leftT, rightT, WT, nrm, output_b, out);
}
